// OffsetAttention_77592879170076
// MI455X (gfx1250) — matrix operand order repaired and checked by reading, not compiled, not run on hardware
//
#include <hip/hip_runtime.h>
#include <hip/hip_bf16.h>

// Offset attention (Point Transformer style), B=4, C=128, N=4096, fp32 in/out.
// flash-attention with v_wmma_f32_16x16x32_f16, async global->LDS double
// buffering (ASYNCcnt) when the toolchain exposes the gfx1250 builtins.
//
// Workspace: 3 * B*N*C * 2B (q_t, k_t, vp) = 12 MiB
//            + B*C*N * 4B (newf)           =  8 MiB
//            + 2*C*4B (stats)              ~ 20 MiB total.

#define B_SZ  4
#define C_DIM 128
#define N_PTS 4096
#define BN_EPS 1e-5f
#define QK_SCALE 0.08838834764831845f   // 1/sqrt(128)

typedef __attribute__((ext_vector_type(16))) _Float16 v16h;
typedef __attribute__((ext_vector_type(8)))  _Float16 v8h;
typedef __attribute__((ext_vector_type(8)))  float    v8f;
typedef int v4i_t __attribute__((vector_size(16)));

#if defined(__has_builtin)
#if __has_builtin(__builtin_amdgcn_global_load_async_to_lds_b128) && \
    __has_builtin(__builtin_amdgcn_s_wait_asynccnt)
#define USE_ASYNC_LDS 1
#endif
#endif
#ifndef USE_ASYNC_LDS
#define USE_ASYNC_LDS 0
#endif

#if USE_ASYNC_LDS
typedef __attribute__((address_space(1))) v4i_t GlobV4;
typedef __attribute__((address_space(3))) v4i_t LdsV4;
// 16-byte async copy global -> LDS (tracked by ASYNCcnt, no VGPR staging).
__device__ __forceinline__ void async_cp16(void* lds_dst, const void* gsrc) {
  __builtin_amdgcn_global_load_async_to_lds_b128(
      (GlobV4*)(uintptr_t)gsrc,
      (LdsV4*)(uint32_t)(uintptr_t)lds_dst, 0, 0);
}
#endif

__device__ __forceinline__ v16h cat16(v8h lo, v8h hi) {
  v16h r;
#pragma unroll
  for (int i = 0; i < 8; ++i) { r[i] = lo[i]; r[8 + i] = hi[i]; }
  return r;
}

// -------------------------------------------------------------------------
// Stage 1: projections.
//   which=0: q_t[b][n][c] = (Wq f + bq) * QK_SCALE      (layout (B,N,C) f16)
//   which=1: k_t[b][n][c] =  Wk f + bk                  (layout (B,N,C) f16)
//   which=2: vp [b][c][n] =  Wv f + bv + Wp xyz^T + bp  (layout (B,C,N) f16)
// -------------------------------------------------------------------------
__global__ __launch_bounds__(256) void project_kernel(
    const float* __restrict__ feat, const float* __restrict__ xyz,
    const float* __restrict__ Wq, const float* __restrict__ bq,
    const float* __restrict__ Wk, const float* __restrict__ bk,
    const float* __restrict__ Wv, const float* __restrict__ bv,
    const float* __restrict__ Wp, const float* __restrict__ bp,
    _Float16* __restrict__ qt, _Float16* __restrict__ kt,
    _Float16* __restrict__ vp) {
  const int n = blockIdx.x * 256 + threadIdx.x;
  const int c = blockIdx.y;
  const int b = blockIdx.z / 3;
  const int which = blockIdx.z % 3;

  const float* W;
  const float* bias;
  if (which == 0)      { W = Wq; bias = bq; }
  else if (which == 1) { W = Wk; bias = bk; }
  else                 { W = Wv; bias = bv; }

  const float* frow = feat + (size_t)b * C_DIM * N_PTS + n;
  const float* wrow = W + c * C_DIM;   // uniform across block -> scalar loads
  float acc = bias[c];
#pragma unroll 8
  for (int cc = 0; cc < C_DIM; ++cc)
    acc += wrow[cc] * frow[(size_t)cc * N_PTS];

  if (which == 0) {
    qt[((size_t)b * N_PTS + n) * C_DIM + c] = (_Float16)(acc * QK_SCALE);
  } else if (which == 1) {
    kt[((size_t)b * N_PTS + n) * C_DIM + c] = (_Float16)acc;
  } else {
    const float* p = xyz + ((size_t)b * N_PTS + n) * 3;
    acc += Wp[c * 3 + 0] * p[0] + Wp[c * 3 + 1] * p[1] + Wp[c * 3 + 2] * p[2] +
           bp[c];
    vp[((size_t)b * C_DIM + c) * N_PTS + n] = (_Float16)acc;
  }
}

// -------------------------------------------------------------------------
// Stage 2: flash attention. One block = one batch x 64 query rows.
// 4 waves x 16 rows each; 256 blocks total. Keys/values stream through a
// double-buffered LDS tile (async copy when available).
// newf[b][c][m] = (P/l) . vp  +  feat[b][c][m]
// -------------------------------------------------------------------------
__device__ __forceinline__ void load_tiles(
    const _Float16* __restrict__ kt, const _Float16* __restrict__ vp,
    _Float16* k_buf, _Float16* vp_buf, int b, int n0, int tid) {
  const _Float16* ksrc = kt + ((size_t)b * N_PTS + n0) * C_DIM;  // contiguous
#if USE_ASYNC_LDS
#pragma unroll
  for (int i = 0; i < 4; ++i) {
    const int ch = tid + i * 128;  // 512 chunks of 16B
    async_cp16(k_buf + ch * 8, ksrc + ch * 8);
  }
#pragma unroll
  for (int i = 0; i < 4; ++i) {
    const int ch = tid + i * 128;
    const int c = ch >> 2, part = ch & 3;
    async_cp16(vp_buf + c * 32 + part * 8,
               vp + ((size_t)b * C_DIM + c) * N_PTS + n0 + part * 8);
  }
#else
#pragma unroll
  for (int i = 0; i < 4; ++i) {
    const int ch = tid + i * 128;
    *(v8h*)(k_buf + ch * 8) = *(const v8h*)(ksrc + ch * 8);
  }
#pragma unroll
  for (int i = 0; i < 4; ++i) {
    const int ch = tid + i * 128;
    const int c = ch >> 2, part = ch & 3;
    *(v8h*)(vp_buf + c * 32 + part * 8) =
        *(const v8h*)(vp + ((size_t)b * C_DIM + c) * N_PTS + n0 + part * 8);
  }
#endif
}

__global__ __launch_bounds__(128) void flash_attn_kernel(
    const _Float16* __restrict__ qt, const _Float16* __restrict__ kt,
    const _Float16* __restrict__ vp, const float* __restrict__ feat,
    float* __restrict__ newf) {
  __shared__ __align__(16) _Float16 lds_k[2][32 * C_DIM];   // [n_local][c]
  __shared__ __align__(16) _Float16 lds_vp[2][C_DIM * 32];  // [c][n_local]
  __shared__ __align__(16) _Float16 lds_p[4][16 * 32];      // per wave [m][n]

  const int b    = blockIdx.y;
  const int tid  = threadIdx.x;
  const int wave = tid >> 5;
  const int lane = tid & 31;
  const int hl   = lane & 15;       // index within 16-lane half
  const bool hi_half = (lane >= 16);
  const int m0   = blockIdx.x * 64 + wave * 16;

  // ---- Preload Q A-fragments (4 c-chunks of 32).
  // A 16x32, this code's k order: lanes 0-15 row=lane, K=c0+0..15;
  //                     lanes 16-31 row=lane-16, K=c0+16..31.
  const int koffA = hi_half ? 16 : 0;
  const _Float16* qrow = qt + ((size_t)b * N_PTS + (m0 + hl)) * C_DIM;
  v16h aq[4];
#pragma unroll
  for (int ct = 0; ct < 4; ++ct) {
    const int c0 = ct * 32;
    aq[ct] = cat16(*(const v8h*)(qrow + c0 + koffA),
                   *(const v8h*)(qrow + c0 + koffA + 8));
  }

  v8f o[8];
#pragma unroll
  for (int t = 0; t < 8; ++t) o[t] = (v8f)(0.0f);
  float m_run[8], l_run[8];
#pragma unroll
  for (int r = 0; r < 8; ++r) { m_run[r] = -1e30f; l_run[r] = 0.0f; }

  // B 32x16, same k order as A: lanes 0-15 col=lane, K=0..15;
  //                     lanes 16-31 col=lane-16, K=16..31.
  const int coffB = hi_half ? 16 : 0;
  _Float16* pbuf = lds_p[wave];

  // Prologue: fetch chunk 0.
  load_tiles(kt, vp, lds_k[0], lds_vp[0], b, 0, tid);
#if USE_ASYNC_LDS
  __builtin_amdgcn_s_wait_asynccnt(0);
#endif
  __syncthreads();

  int buf = 0;
  for (int n0 = 0; n0 < N_PTS; n0 += 32, buf ^= 1) {
    const _Float16* k_cur = lds_k[buf];
    const _Float16* vp_cur = lds_vp[buf];

    // Prefetch next chunk into the other buffer while we compute.
    if (n0 + 32 < N_PTS)
      load_tiles(kt, vp, lds_k[buf ^ 1], lds_vp[buf ^ 1], b, n0 + 32, tid);

    // ---- Preload all S-phase B fragments, then a clean run of 8 WMMAs.
    v16h bb0[4], bb1[4];
    const _Float16* k0 = k_cur + hl * C_DIM;         // col n = hl
    const _Float16* k1 = k_cur + (hl + 16) * C_DIM;  // col n = hl+16
#pragma unroll
    for (int ct = 0; ct < 4; ++ct) {
      const int c0 = ct * 32 + coffB;
      bb0[ct] = cat16(*(const v8h*)(k0 + c0), *(const v8h*)(k0 + c0 + 8));
      bb1[ct] = cat16(*(const v8h*)(k1 + c0), *(const v8h*)(k1 + c0 + 8));
    }
    v8f s0 = (v8f)(0.0f), s1 = (v8f)(0.0f);
#pragma unroll
    for (int ct = 0; ct < 4; ++ct) {
      s0 = __builtin_amdgcn_wmma_f32_16x16x32_f16(false, aq[ct], false,
                                                  bb0[ct], (short)0, s0,
                                                  false, false);
      s1 = __builtin_amdgcn_wmma_f32_16x16x32_f16(false, aq[ct], false,
                                                  bb1[ct], (short)0, s1,
                                                  false, false);
    }

    // ---- Online softmax. D layout: VGPR r -> row r (lanes 0-15) / r+8
    // (lanes 16-31); reductions stay within 16-lane halves (xor 8,4,2,1).
    float corr[8];
#pragma unroll
    for (int r = 0; r < 8; ++r) {
      float v = fmaxf(s0[r], s1[r]);
#pragma unroll
      for (int off = 8; off >= 1; off >>= 1)
        v = fmaxf(v, __shfl_xor(v, off, 32));
      const float mnew = fmaxf(m_run[r], v);
      corr[r] = __expf(m_run[r] - mnew);
      const float p0 = __expf(s0[r] - mnew);
      const float p1 = __expf(s1[r] - mnew);
      s0[r] = p0; s1[r] = p1;
      float rs = p0 + p1;
#pragma unroll
      for (int off = 8; off >= 1; off >>= 1) rs += __shfl_xor(rs, off, 32);
      l_run[r] = l_run[r] * corr[r] + rs;
      m_run[r] = mnew;
    }
#pragma unroll
    for (int t = 0; t < 8; ++t) {
#pragma unroll
      for (int r = 0; r < 8; ++r) o[t][r] *= corr[r];
    }

    // ---- Preload all O-phase B fragments (independent of P repack).
    v16h vb[8];
#pragma unroll
    for (int ct = 0; ct < 8; ++ct) {
      const _Float16* vrow = vp_cur + (ct * 16 + hl) * 32 + coffB;
      vb[ct] = cat16(*(const v8h*)(vrow), *(const v8h*)(vrow + 8));
    }

    // ---- Repack P (D layout) -> A layout via per-wave LDS tile.
#pragma unroll
    for (int r = 0; r < 8; ++r) {
      const int row = r + (hi_half ? 8 : 0);
      pbuf[row * 32 + hl] = (_Float16)s0[r];
      pbuf[row * 32 + hl + 16] = (_Float16)s1[r];
    }
#if __has_builtin(__builtin_amdgcn_s_wait_dscnt)
    __builtin_amdgcn_s_wait_dscnt(0);  // wave-private tile: DS is in-order
#endif
    const _Float16* prow = pbuf + hl * 32 + koffA;
    const v16h pa = cat16(*(const v8h*)(prow), *(const v8h*)(prow + 8));

    // ---- O += P . VP^T : 8 independent WMMAs, back to back.
#pragma unroll
    for (int ct = 0; ct < 8; ++ct) {
      o[ct] = __builtin_amdgcn_wmma_f32_16x16x32_f16(false, pa, false, vb[ct],
                                                     (short)0, o[ct], false,
                                                     false);
    }

#if USE_ASYNC_LDS
    __builtin_amdgcn_s_wait_asynccnt(0);  // prefetch landed
#endif
    __syncthreads();  // all waves done with cur; next buffer visible
  }

  // ---- Epilogue: normalize by l, add residual features, store fp32.
  float inv_l[8];
#pragma unroll
  for (int r = 0; r < 8; ++r) inv_l[r] = 1.0f / fmaxf(l_run[r], 1e-30f);
#pragma unroll
  for (int ct = 0; ct < 8; ++ct) {
#pragma unroll
    for (int r = 0; r < 8; ++r) {
      const int c = ct * 16 + hl;
      const int m = m0 + r + (hi_half ? 8 : 0);
      const size_t idx = ((size_t)b * C_DIM + c) * N_PTS + m;
      newf[idx] = o[ct][r] * inv_l[r] + feat[idx];
    }
  }
}

// -------------------------------------------------------------------------
// Stage 3: per-channel batch stats over (B, N).
// -------------------------------------------------------------------------
__global__ __launch_bounds__(256) void bn_stats_kernel(
    const float* __restrict__ newf, float* __restrict__ mean,
    float* __restrict__ rstd) {
  const int c = blockIdx.x;
  const int tid = threadIdx.x;
  float s = 0.0f, s2 = 0.0f;
  for (int i = tid; i < B_SZ * N_PTS; i += 256) {
    const int b = i >> 12, n = i & (N_PTS - 1);
    const float v = newf[((size_t)b * C_DIM + c) * N_PTS + n];
    s += v; s2 += v * v;
  }
  __shared__ float sh0[256], sh1[256];
  sh0[tid] = s; sh1[tid] = s2;
  __syncthreads();
  for (int st = 128; st > 0; st >>= 1) {
    if (tid < st) { sh0[tid] += sh0[tid + st]; sh1[tid] += sh1[tid + st]; }
    __syncthreads();
  }
  if (tid == 0) {
    const float inv_m = 1.0f / (float)(B_SZ * N_PTS);
    const float mu = sh0[0] * inv_m;
    const float var = sh1[0] * inv_m - mu * mu;
    mean[c] = mu;
    rstd[c] = rsqrtf(var + BN_EPS);
  }
}

// -------------------------------------------------------------------------
// Stage 4: normalize + affine + relu.
// -------------------------------------------------------------------------
__global__ __launch_bounds__(256) void bn_apply_kernel(
    const float* __restrict__ newf, const float* __restrict__ mean,
    const float* __restrict__ rstd, const float* __restrict__ gamma,
    const float* __restrict__ beta, float* __restrict__ out) {
  const size_t i = (size_t)blockIdx.x * 256 + threadIdx.x;
  const int c = (int)((i / N_PTS) % C_DIM);
  const float v = (newf[i] - mean[c]) * rstd[c] * gamma[c] + beta[c];
  out[i] = fmaxf(v, 0.0f);
}

// -------------------------------------------------------------------------
extern "C" void kernel_launch(void* const* d_in, const int* in_sizes, int n_in,
                              void* d_out, int out_size, void* d_ws,
                              size_t ws_size, hipStream_t stream) {
  const float* xyz   = (const float*)d_in[0];
  const float* feat  = (const float*)d_in[1];
  const float* Wq    = (const float*)d_in[2];
  const float* bq    = (const float*)d_in[3];
  const float* Wk    = (const float*)d_in[4];
  const float* bk    = (const float*)d_in[5];
  const float* Wv    = (const float*)d_in[6];
  const float* bv    = (const float*)d_in[7];
  const float* Wp    = (const float*)d_in[8];
  const float* bp    = (const float*)d_in[9];
  const float* gamma = (const float*)d_in[10];
  const float* beta  = (const float*)d_in[11];
  float* out = (float*)d_out;

  const size_t elems = (size_t)B_SZ * N_PTS * C_DIM;  // 2,097,152
  _Float16* qt = (_Float16*)d_ws;
  _Float16* kt = qt + elems;
  _Float16* vp = kt + elems;
  float* newf  = (float*)(vp + elems);
  float* mean  = newf + elems;
  float* rstd  = mean + C_DIM;

  project_kernel<<<dim3(N_PTS / 256, C_DIM, B_SZ * 3), 256, 0, stream>>>(
      feat, xyz, Wq, bq, Wk, bk, Wv, bv, Wp, bp, qt, kt, vp);

  flash_attn_kernel<<<dim3(N_PTS / 64, B_SZ), 128, 0, stream>>>(
      qt, kt, vp, feat, newf);

  bn_stats_kernel<<<C_DIM, 256, 0, stream>>>(newf, mean, rstd);

  bn_apply_kernel<<<(int)(elems / 256), 256, 0, stream>>>(newf, mean, rstd,
                                                          gamma, beta, out);
}
